// Block_90194313216506
// MI455X (gfx1250) — matrix operand order repaired and checked by reading, not compiled, not run on hardware
//
#include <hip/hip_runtime.h>

// ---------------------------------------------------------------------------
// Types (trivially-copyable vectors so they can live inside unions)
// ---------------------------------------------------------------------------
typedef __attribute__((ext_vector_type(16))) __bf16        v16bf;
typedef __attribute__((ext_vector_type(8)))  float         v8f;
typedef __attribute__((ext_vector_type(4)))  unsigned int  v4u;

union FragAB { v16bf v; v4u q[2]; };         // 16 bf16 = 32 bytes = 8 VGPRs

__device__ __forceinline__ unsigned short f2bf(float f) {
  unsigned int u = __float_as_uint(f);
  u += 0x7FFFu + ((u >> 16) & 1u);           // round-to-nearest-even
  return (unsigned short)(u >> 16);
}

// Async global->LDS DMA (CDNA5, ASYNCcnt-tracked). LDS operand is the low 32
// bits of the generic shared-memory pointer (flat addr truncation, ISA 10.2).
__device__ __forceinline__ void async_load_b128(const void* gptr, void* lptr) {
  unsigned int lo = (unsigned int)(unsigned long long)lptr;
  asm volatile("global_load_async_to_lds_b128 %0, %1, off"
               :: "v"(lo), "v"((unsigned long long)gptr) : "memory");
}
__device__ __forceinline__ void wait_async_le(int n) {
  if (n == 0)      asm volatile("s_wait_asynccnt 0x0" ::: "memory");
  else if (n == 2) asm volatile("s_wait_asynccnt 0x2" ::: "memory");
  else             asm volatile("s_wait_asynccnt 0x4" ::: "memory");
}

// ---------------------------------------------------------------------------
// LayerNorm -> bf16
// ---------------------------------------------------------------------------
__global__ __launch_bounds__(256) void ln_bf16_kernel(
    const float* __restrict__ x, const float* __restrict__ g,
    const float* __restrict__ be, unsigned short* __restrict__ out, int E)
{
  const int row = blockIdx.x;
  const int tid = threadIdx.x;
  const float* xr = x + (long long)row * E;
  float s1 = 0.f, s2 = 0.f;
  for (int j = tid; j < E; j += 256) { float v = xr[j]; s1 += v; s2 += v * v; }
  __shared__ float r1[256], r2[256];
  r1[tid] = s1; r2[tid] = s2;
  __syncthreads();
  for (int off = 128; off > 0; off >>= 1) {
    if (tid < off) { r1[tid] += r1[tid + off]; r2[tid] += r2[tid + off]; }
    __syncthreads();
  }
  const float mu  = r1[0] / E;
  const float var = r2[0] / E - mu * mu;
  const float inv = rsqrtf(var + 1e-5f);
  unsigned short* orow = out + (long long)row * E;
  for (int j = tid; j < E; j += 256)
    orow[j] = f2bf((xr[j] - mu) * inv * g[j] + be[j]);
}

// ---------------------------------------------------------------------------
// f32 [K,N] -> bf16 [N,K] transpose (weights), batched via grid.z
// ---------------------------------------------------------------------------
__global__ __launch_bounds__(256) void transpose_f32_bf16_kernel(
    const float* __restrict__ in, unsigned short* __restrict__ out,
    int K, int N, long long inStride, long long outStride)
{
  __shared__ float t[32][33];
  const long long ib = (long long)blockIdx.z * inStride;
  const long long ob = (long long)blockIdx.z * outStride;
  const int k0 = blockIdx.y * 32, n0 = blockIdx.x * 32;
  const int tx = threadIdx.x, ty = threadIdx.y;   // blockDim = (32, 8)
#pragma unroll
  for (int i = 0; i < 4; ++i)
    t[ty + i * 8][tx] = in[ib + (long long)(k0 + ty + i * 8) * N + n0 + tx];
  __syncthreads();
#pragma unroll
  for (int i = 0; i < 4; ++i)
    out[ob + (long long)(n0 + ty + i * 8) * K + k0 + tx] = f2bf(t[tx][ty + i * 8]);
}

// ---------------------------------------------------------------------------
// Softmax over rows of Sc (f32), faithful post-softmax masking, bf16 out
// ---------------------------------------------------------------------------
__global__ __launch_bounds__(256) void softmax_mask_kernel(
    const float* __restrict__ Sc, const int* __restrict__ mask,
    unsigned short* __restrict__ P, int S, int HS)
{
  const long long row = blockIdx.x;               // B*H*S rows
  const int b = (int)(row / HS);
  const float* xr = Sc + row * S;
  unsigned short* pr = P + row * S;
  const int tid = threadIdx.x;
  __shared__ float red[256];

  float lmax = -3.4e38f;
  for (int j = tid; j < S; j += 256) lmax = fmaxf(lmax, xr[j]);
  red[tid] = lmax; __syncthreads();
  for (int off = 128; off > 0; off >>= 1) {
    if (tid < off) red[tid] = fmaxf(red[tid], red[tid + off]);
    __syncthreads();
  }
  const float gmax = red[0];
  __syncthreads();

  float lsum = 0.f;
  for (int j = tid; j < S; j += 256) lsum += __expf(xr[j] - gmax);
  red[tid] = lsum; __syncthreads();
  for (int off = 128; off > 0; off >>= 1) {
    if (tid < off) red[tid] += red[tid + off];
    __syncthreads();
  }
  const float inv = 1.f / red[0];
  const int* mrow = mask + (long long)b * S;
  for (int j = tid; j < S; j += 256) {
    float p = __expf(xr[j] - gmax) * inv;
    if (mrow[j] == 0) p = -1e9f;                  // reference masks AFTER softmax
    pr[j] = f2bf(p);
  }
}

// ---------------------------------------------------------------------------
// bf16 WMMA GEMM, 128x128x32 block tile, 8 wave32s (each 64x32 -> 8 C frags),
// double-buffered LDS with async global->LDS DMA staging.
//   A:  [M,K] bf16 row-major
//   B:  BT ? [N,K] bf16 (pre-transposed)  :  [K,N] bf16 (transposed in LDS)
// ---------------------------------------------------------------------------
enum { EPI_BIAS_BF16 = 0, EPI_SCALE_F32, EPI_PLAIN_BF16, EPI_BIAS_RES_F32,
       EPI_RELU_BIAS_BF16 };

template<int EPI, bool BT>
__global__ __launch_bounds__(256, 1) void gemm_bf16_kernel(
    const unsigned short* __restrict__ A, long long sAo, long long sAi, int lda,
    const unsigned short* __restrict__ Bm, long long sBo, long long sBi, int ldb,
    const float* __restrict__ bias, long long sBias,
    float* __restrict__ outF, unsigned short* __restrict__ outB,
    long long sCo, long long sCi, int ldc,
    const float* __restrict__ resid,
    int M, int N, int K, int batchH, float scale)
{
  (void)M;
  constexpr int BMt = 128, BNt = 128, BKt = 32;
  constexpr int LDSS = 40;                       // padded row: 80B, 16B aligned
  __shared__ unsigned short As[2][BMt * LDSS];   // 2 x 10 KB
  __shared__ unsigned short Bs[2][BNt * LDSS];   // 2 x 10 KB

  const int z  = blockIdx.z;
  const int zb = z / batchH, zh = z - zb * batchH;
  const unsigned short* Ab = A  + (long long)zb * sAo + (long long)zh * sAi;
  const unsigned short* Bb = Bm + (long long)zb * sBo + (long long)zh * sBi;

  const int tid  = threadIdx.x;
  const int wid  = tid >> 5;                     // wave32
  const int lane = tid & 31;
  const int wm   = wid >> 2;                     // 0..1  (M direction)
  const int wn   = wid & 3;                      // 0..3  (N direction)
  const int half = lane >> 4;
  const int l16  = lane & 15;

  const int rowBlk = blockIdx.y * BMt;
  const int colBlk = blockIdx.x * BNt;

  // per-thread staging coordinates (512 16B-chunks per 128x32 tile)
  const int aM0 = tid >> 2,        aK0 = (tid & 3) * 8;        // chunk tid
  const int aM1 = (tid + 256) >> 2, aK1 = ((tid + 256) & 3) * 8;
  const int bK0 = tid >> 4,        bN0 = (tid & 15) * 8;       // !BT staging
  const int bK1 = (tid + 256) >> 4, bN1 = ((tid + 256) & 15) * 8;

  // number of async ops issued per thread per stage (A always async; B async
  // only when pre-transposed)
  constexpr int PER_STAGE = BT ? 4 : 2;

  // ---- stage one K-tile into LDS buffer `buf` -----------------------------
  auto stage = [&](int k0, int buf) {
    async_load_b128(Ab + (long long)(rowBlk + aM0) * lda + k0 + aK0,
                    &As[buf][aM0 * LDSS + aK0]);
    async_load_b128(Ab + (long long)(rowBlk + aM1) * lda + k0 + aK1,
                    &As[buf][aM1 * LDSS + aK1]);
    if (BT) {
      async_load_b128(Bb + (long long)(colBlk + aM0) * ldb + k0 + aK0,
                      &Bs[buf][aM0 * LDSS + aK0]);
      async_load_b128(Bb + (long long)(colBlk + aM1) * ldb + k0 + aK1,
                      &Bs[buf][aM1 * LDSS + aK1]);
    } else {                                     // row-major B: LDS transpose
      union { v4u q; unsigned short s[8]; } u0, u1;
      u0.q = *(const v4u*)(Bb + (long long)(k0 + bK0) * ldb + colBlk + bN0);
      u1.q = *(const v4u*)(Bb + (long long)(k0 + bK1) * ldb + colBlk + bN1);
#pragma unroll
      for (int i = 0; i < 8; ++i) Bs[buf][(bN0 + i) * LDSS + bK0] = u0.s[i];
#pragma unroll
      for (int i = 0; i < 8; ++i) Bs[buf][(bN1 + i) * LDSS + bK1] = u1.s[i];
    }
  };

  v8f acc[4][2];
#pragma unroll
  for (int mi = 0; mi < 4; ++mi)
#pragma unroll
    for (int ni = 0; ni < 2; ++ni)
#pragma unroll
      for (int r = 0; r < 8; ++r) acc[mi][ni][r] = 0.f;

  const int kTiles = K / BKt;
  stage(0, 0);                                   // prologue

  for (int kt = 0; kt < kTiles; ++kt) {
    const int buf = kt & 1;
    if (kt + 1 < kTiles) {
      stage((kt + 1) * BKt, buf ^ 1);            // DMA next tile (other buffer)
      wait_async_le(PER_STAGE);                  // in-order => current tile done
      // prefetch k+2 into GL2 while the matrix pipe works
      if (kt + 2 < kTiles) {
        const int k2 = (kt + 2) * BKt;
        __builtin_prefetch(Ab + (long long)(rowBlk + aM0) * lda + k2 + aK0, 0, 1);
        if (BT)
          __builtin_prefetch(Bb + (long long)(colBlk + aM0) * ldb + k2 + aK0, 0, 1);
        else
          __builtin_prefetch(Bb + (long long)(k2 + bK0) * ldb + colBlk + bN0, 0, 1);
      }
    } else {
      wait_async_le(0);
    }
    __syncthreads();                             // tile `buf` visible to all

    // ---- fragments (ISA 7.12.2 layouts) ----------------------------------
    FragAB af[4], bfr[2];
#pragma unroll
    for (int mi = 0; mi < 4; ++mi) {             // A: lane row = l16, K halves
      const int r = (wm * 64 + mi * 16 + l16) * LDSS;
      af[mi].q[0] = *(const v4u*)&As[buf][r + half * 8];       // K 0..15 half
      af[mi].q[1] = *(const v4u*)&As[buf][r + 16 + half * 8];  // K 16..31 half
    }
#pragma unroll
    for (int ni = 0; ni < 2; ++ni) {             // B: lane col = l16, K halves (same as A)
      const int c = (wn * 32 + ni * 16 + l16) * LDSS;
      bfr[ni].q[0] = *(const v4u*)&Bs[buf][c + half * 8];
      bfr[ni].q[1] = *(const v4u*)&Bs[buf][c + 16 + half * 8];
    }
#pragma unroll
    for (int mi = 0; mi < 4; ++mi)
#pragma unroll
      for (int ni = 0; ni < 2; ++ni)
        acc[mi][ni] = __builtin_amdgcn_wmma_f32_16x16x32_bf16(
            false, af[mi].v, false, bfr[ni].v, (short)0, acc[mi][ni],
            false, false);
    __syncthreads();                             // done reading `buf`
  }

  // ---- epilogue: C layout row = r + 8*half, col = l16 ---------------------
  const long long cOff = (long long)zb * sCo + (long long)zh * sCi;
#pragma unroll
  for (int mi = 0; mi < 4; ++mi) {
#pragma unroll
    for (int ni = 0; ni < 2; ++ni) {
#pragma unroll
      for (int r = 0; r < 8; ++r) {
        const int row = rowBlk + wm * 64 + mi * 16 + half * 8 + r;
        const int col = colBlk + wn * 32 + ni * 16 + l16;
        float v = acc[mi][ni][r];
        if (EPI == EPI_SCALE_F32) v *= scale;
        if (EPI == EPI_BIAS_BF16 || EPI == EPI_BIAS_RES_F32 ||
            EPI == EPI_RELU_BIAS_BF16)
          v += bias[(long long)zh * sBias + col];
        if (EPI == EPI_RELU_BIAS_BF16) v = v > 0.f ? v : 0.f;
        if (EPI == EPI_BIAS_RES_F32) v += resid[(long long)row * ldc + col];
        const long long idx = cOff + (long long)row * ldc + col;
        if (EPI == EPI_SCALE_F32 || EPI == EPI_BIAS_RES_F32) outF[idx] = v;
        else                                                  outB[idx] = f2bf(v);
      }
    }
  }
}

// ---------------------------------------------------------------------------
// Host-side orchestration
// ---------------------------------------------------------------------------
extern "C" void kernel_launch(void* const* d_in, const int* in_sizes, int n_in,
                              void* d_out, int out_size, void* d_ws, size_t ws_size,
                              hipStream_t stream)
{
  (void)in_sizes; (void)n_in; (void)out_size; (void)ws_size;
  const float* x    = (const float*)d_in[0];
  const int*   mask = (const int*)  d_in[1];
  const float* Wq   = (const float*)d_in[2];
  const float* bq   = (const float*)d_in[3];
  const float* Wk   = (const float*)d_in[4];
  const float* bk   = (const float*)d_in[5];
  const float* Wv   = (const float*)d_in[6];
  const float* bv   = (const float*)d_in[7];
  const float* Wo   = (const float*)d_in[8];
  const float* bo   = (const float*)d_in[9];
  const float* W1   = (const float*)d_in[10];
  const float* b1   = (const float*)d_in[11];
  const float* W2   = (const float*)d_in[12];
  const float* b2   = (const float*)d_in[13];
  const float* g1   = (const float*)d_in[14];
  const float* be1  = (const float*)d_in[15];
  const float* g2   = (const float*)d_in[16];
  const float* be2  = (const float*)d_in[17];
  float* out = (float*)d_out;

  const long long B = 8, S = 1024, E = 1024, H = 14, D = 1024, F = 4096;
  const long long HD = H * D, BS = B * S;

  char* p = (char*)d_ws;
  auto take = [&](long long bytes) -> char* {
    char* r = p; p += (bytes + 255) & ~255LL; return r;
  };
  unsigned short* hbf  = (unsigned short*)take(BS * E * 2);
  unsigned short* h2bf = (unsigned short*)take(BS * E * 2);
  unsigned short* WqT  = (unsigned short*)take(H * E * D * 2);
  unsigned short* WkT  = (unsigned short*)take(H * E * D * 2);
  unsigned short* WvT  = (unsigned short*)take(H * E * D * 2);
  unsigned short* WoT  = (unsigned short*)take(E * HD * 2);
  unsigned short* W1T  = (unsigned short*)take(F * E * 2);
  unsigned short* W2T  = (unsigned short*)take(E * F * 2);
  unsigned short* Qb   = (unsigned short*)take(B * H * S * D * 2);
  unsigned short* Kb   = (unsigned short*)take(B * H * S * D * 2);
  unsigned short* Vb   = (unsigned short*)take(B * H * S * D * 2);
  float*          Sc   = (float*)         take(B * H * S * S * 4);
  unsigned short* Pb   = (unsigned short*)take(B * H * S * S * 2);
  unsigned short* Ob   = (unsigned short*)take(BS * HD * 2);
  float*          x2   = (float*)         take(BS * E * 4);
  unsigned short* f1   = (unsigned short*)take(BS * F * 2);

  const dim3 tb(32, 8);
  // --- weight conversion + transpose (amortized once per launch) ----------
  transpose_f32_bf16_kernel<<<dim3(D/32, E/32, H), tb, 0, stream>>>(Wq, WqT, E, D, E*D, D*E);
  transpose_f32_bf16_kernel<<<dim3(D/32, E/32, H), tb, 0, stream>>>(Wk, WkT, E, D, E*D, D*E);
  transpose_f32_bf16_kernel<<<dim3(D/32, E/32, H), tb, 0, stream>>>(Wv, WvT, E, D, E*D, D*E);
  transpose_f32_bf16_kernel<<<dim3(E/32, HD/32, 1), tb, 0, stream>>>(Wo, WoT, HD, E, 0, 0);
  transpose_f32_bf16_kernel<<<dim3(F/32, E/32, 1),  tb, 0, stream>>>(W1, W1T, E, F, 0, 0);
  transpose_f32_bf16_kernel<<<dim3(E/32, F/32, 1),  tb, 0, stream>>>(W2, W2T, F, E, 0, 0);

  // --- LN1 -----------------------------------------------------------------
  ln_bf16_kernel<<<(int)BS, 256, 0, stream>>>(x, g1, be1, hbf, (int)E);

  // --- Q/K/V projections: per (b,h) [S,E]x[E,D] ---------------------------
  gemm_bf16_kernel<EPI_BIAS_BF16, true><<<dim3(D/128, S/128, B*H), 256, 0, stream>>>(
      hbf, S*E, 0, (int)E,  WqT, 0, D*E, (int)E,  bq, D,
      nullptr, Qb, H*S*D, S*D, (int)D,  nullptr,
      (int)S, (int)D, (int)E, (int)H, 1.f);
  gemm_bf16_kernel<EPI_BIAS_BF16, true><<<dim3(D/128, S/128, B*H), 256, 0, stream>>>(
      hbf, S*E, 0, (int)E,  WkT, 0, D*E, (int)E,  bk, D,
      nullptr, Kb, H*S*D, S*D, (int)D,  nullptr,
      (int)S, (int)D, (int)E, (int)H, 1.f);
  gemm_bf16_kernel<EPI_BIAS_BF16, true><<<dim3(D/128, S/128, B*H), 256, 0, stream>>>(
      hbf, S*E, 0, (int)E,  WvT, 0, D*E, (int)E,  bv, D,
      nullptr, Vb, H*S*D, S*D, (int)D,  nullptr,
      (int)S, (int)D, (int)E, (int)H, 1.f);

  // --- scores = Q Kᵀ * D^-0.5  (K row-major == pre-transposed B) ----------
  gemm_bf16_kernel<EPI_SCALE_F32, true><<<dim3(S/128, S/128, B*H), 256, 0, stream>>>(
      Qb, S*D, 0, (int)D,  Kb, S*D, 0, (int)D,  nullptr, 0,
      Sc, nullptr, S*S, 0, (int)S,  nullptr,
      (int)S, (int)S, (int)D, 1, 0.03125f /* 1/sqrt(1024) */);

  // --- softmax + post-softmax mask ----------------------------------------
  softmax_mask_kernel<<<(int)(B*H*S), 256, 0, stream>>>(
      Sc, mask, Pb, (int)S, (int)(H*S));

  // --- O = A V (B row-major -> LDS transpose path), scatter into [B,S,H*D]
  gemm_bf16_kernel<EPI_PLAIN_BF16, false><<<dim3(D/128, S/128, B*H), 256, 0, stream>>>(
      Pb, H*S*S, S*S, (int)S,  Vb, H*S*D, S*D, (int)D,  nullptr, 0,
      nullptr, Ob, S*HD, D, (int)HD,  nullptr,
      (int)S, (int)D, (int)S, (int)H, 1.f);

  // --- x2 = x + O @ Wo + bo ------------------------------------------------
  gemm_bf16_kernel<EPI_BIAS_RES_F32, true><<<dim3(E/128, BS/128, 1), 256, 0, stream>>>(
      Ob, 0, 0, (int)HD,  WoT, 0, 0, (int)HD,  bo, 0,
      x2, nullptr, 0, 0, (int)E,  x,
      (int)BS, (int)E, (int)HD, 1, 1.f);

  // --- LN2 -----------------------------------------------------------------
  ln_bf16_kernel<<<(int)BS, 256, 0, stream>>>(x2, g2, be2, h2bf, (int)E);

  // --- FFN -----------------------------------------------------------------
  gemm_bf16_kernel<EPI_RELU_BIAS_BF16, true><<<dim3(F/128, BS/128, 1), 256, 0, stream>>>(
      h2bf, 0, 0, (int)E,  W1T, 0, 0, (int)E,  b1, 0,
      nullptr, f1, 0, 0, (int)F,  nullptr,
      (int)BS, (int)F, (int)E, 1, 1.f);
  gemm_bf16_kernel<EPI_BIAS_RES_F32, true><<<dim3(E/128, BS/128, 1), 256, 0, stream>>>(
      f1, 0, 0, (int)F,  W2T, 0, 0, (int)F,  b2, 0,
      out, nullptr, 0, 0, (int)E,  x2,
      (int)BS, (int)E, (int)F, 1, 1.f);
}
